// Cross_Mamba_Attention_SSMall_71545565217073
// MI455X (gfx1250) — matrix operand order repaired and checked by reading, not compiled, not run on hardware
//
#include <hip/hip_runtime.h>

// ---------------------------------------------------------------------------
// Cross-Mamba selective scan for MI455X (gfx1250, wave32).
//   K1: input projections xd = x @ Wxp^T via v_wmma_f32_16x16x32_f16 (f32 acc)
//   K2: segmented scan pass 1 -> per-segment end states + dA products
//   K3: sequential carry chain across segments (tiny)
//   K4: segmented scan pass 2 with carries, cross-stream C, fused LayerNorm
// Staging of broadcast rows uses GLOBAL_LOAD_ASYNC_TO_LDS (ASYNCcnt) when the
// toolchain exposes the builtins, else a plain LDS copy.
// ---------------------------------------------------------------------------

#define B_   4
#define L_   2048
#define DI   384
#define N_   16
#define R_   12
#define E_   44            // R_ + 2*N_
#define XST  48            // padded xd row stride (3 aligned 16-col WMMA tiles)
#define P_   16            // segments of L
#define SEG  128           // L_ / P_
#define CH   64            // LDS staging chunk (timesteps)
#define LNEPS 1e-5f

typedef __attribute__((ext_vector_type(16))) _Float16 v16h;
typedef __attribute__((ext_vector_type(8)))  float    v8f;

// float elements per workspace region
#define XD_ELEMS   (B_ * L_ * XST)          // 393216
#define CAR_ELEMS  (2 * B_ * P_ * DI * N_)  // 786432

#if defined(__AMDGCN__) && __has_builtin(__builtin_amdgcn_global_load_async_to_lds_b32) && __has_builtin(__builtin_amdgcn_s_wait_asynccnt)
#define USE_ASYNC_LDS 1
typedef __attribute__((address_space(1))) int* as1_int;
typedef __attribute__((address_space(3))) int* as3_int;

// global float* (maybe const) + LDS float*  ->  async b32 copy into LDS
__device__ __forceinline__ void async_cp_b32(const float* g, float* l) {
    __builtin_amdgcn_global_load_async_to_lds_b32(
        (as1_int)(int*)const_cast<float*>(g),   // addrspacecast generic->AS(1)
        (as3_int)(int*)l,                        // addrspacecast generic->AS(3)
        0, 0);
}
#else
#define USE_ASYNC_LDS 0
#endif

__device__ __forceinline__ float softplus_f(float x) {
    return (x > 20.f) ? x : log1pf(__expf(x));
}

// A-fragment: lane m = lane&15 is row M; lanes 0-15 carry K {0..7,16..23},
// lanes 16-31 carry K {8..15,24..31}.  p points at (row, kchunk+kbase).
__device__ __forceinline__ v16h packA(const float* __restrict__ p) {
    v16h v;
#pragma unroll
    for (int i = 0; i < 8; ++i) v[i]     = (_Float16)p[i];
#pragma unroll
    for (int i = 0; i < 8; ++i) v[8 + i] = (_Float16)p[16 + i];
    return v;
}

// B-fragment: lane n = lane&15 is column; lanes 0-15 hold K {0..7,16..23},
// lanes 16-31 hold K {8..15,24..31}.  p points at (wrow, kchunk+bbase).
__device__ __forceinline__ v16h packB(const float* __restrict__ p) {
    v16h v;
#pragma unroll
    for (int i = 0; i < 16; ++i) v[i] = (_Float16)p[(i < 8) ? i : (8 + i)];
    return v;
}

// ---------------------------------------------------------------------------
// K1: xd[s] = x[s] @ Wxp[s]^T   (rows = B*L = 8192, cols = 44 -> 3 tiles, K=384)
// one wave per 16x16 tile; 2 streams * 512 row tiles * 3 col tiles = 3072 waves.
// Weight rows >= 44 are clamped to row 43: the garbage results land in the
// dead padding columns 44..47 of the stride-48 xd layout (never read), which
// keeps EXEC uniform and the inner loop branch-free.
// ---------------------------------------------------------------------------
__global__ void __launch_bounds__(128)
xmamba_proj(const float* __restrict__ x1, const float* __restrict__ x2,
            const float* __restrict__ W1, const float* __restrict__ W2,
            float* __restrict__ xd1, float* __restrict__ xd2) {
    const int wid  = (blockIdx.x * blockDim.x + threadIdx.x) >> 5;
    const int lane = threadIdx.x & 31;
    const int ROWT = (B_ * L_) / 16;   // 512
    const int JOBS = 2 * ROWT * 3;
    if (wid >= JOBS) return;           // uniform per wave: EXEC stays all-1

    const int s   = wid / (ROWT * 3);
    const int rem = wid % (ROWT * 3);
    const int rt  = rem / 3;
    const int ct  = rem % 3;

    const float* __restrict__ X  = s ? x2  : x1;
    const float* __restrict__ W  = s ? W2  : W1;
    float*       __restrict__ XD = s ? xd2 : xd1;

    const int m     = lane & 15;
    const int hi    = lane >> 4;
    const int kbase = hi * 8;     // A-frag K base
    const int bbase = hi * 8;     // B-frag K base

    const float* arow = X + (size_t)(rt * 16 + m) * DI;
    int wr = ct * 16 + m;                     // weight row == output column
    if (wr > E_ - 1) wr = E_ - 1;             // clamp: no divergent branch
    const float* brow = W + (size_t)wr * DI;

    v8f c = {};
#pragma unroll
    for (int kk = 0; kk < DI; kk += 32) {
        v16h a = packA(arow + kk + kbase);
        v16h b = packB(brow + kk + bbase);
        c = __builtin_amdgcn_wmma_f32_16x16x32_f16(
                false, a, false, b, (short)0, c, false, false);
    }

    // D layout: VGPR v -> M = v + 8*hi, N = lane&15
    float* drow = XD + (size_t)(rt * 16 + hi * 8) * XST + ct * 16 + m;
#pragma unroll
    for (int v = 0; v < 8; ++v) drow[(size_t)v * XST] = c[v];
}

// ---------------------------------------------------------------------------
// K2: pass 1 — per-segment local scan with x0 = 0.
// block = (stream, batch, segment); 384 threads, thread = channel d.
// ---------------------------------------------------------------------------
__global__ void __launch_bounds__(DI)
xmamba_scan1(const float* __restrict__ x1, const float* __restrict__ x2,
             const float* __restrict__ dtW1, const float* __restrict__ dtW2,
             const float* __restrict__ dtb1, const float* __restrict__ dtb2,
             const float* __restrict__ Al1,  const float* __restrict__ Al2,
             const float* __restrict__ xd1,  const float* __restrict__ xd2,
             float* __restrict__ seg_end, float* __restrict__ seg_prod) {
    const int bid = blockIdx.x;                 // (s*B + b)*P + p
    const int p   = bid % P_;
    const int b   = (bid / P_) % B_;
    const int s   = bid / (P_ * B_);
    const int d   = threadIdx.x;

    const float* __restrict__ X   = s ? x2  : x1;
    const float* __restrict__ XD  = s ? xd2 : xd1;
    const float* __restrict__ wdt = (s ? dtW2 : dtW1) + (size_t)d * R_;
    const float  dbias            = (s ? dtb2 : dtb1)[d];
    const float* __restrict__ alg = (s ? Al2 : Al1) + (size_t)d * N_;

    float A[N_], xs[N_], pr[N_], w[R_];
#pragma unroll
    for (int n = 0; n < N_; ++n) { A[n] = -__expf(alg[n]); xs[n] = 0.f; pr[n] = 1.f; }
#pragma unroll
    for (int r = 0; r < R_; ++r) w[r] = wdt[r];

    __shared__ float sb[CH * 28];               // dt(12) + B(16) per step
    const int l0 = p * SEG;

    for (int c0 = 0; c0 < SEG; c0 += CH) {
        __syncthreads();
#if USE_ASYNC_LDS
        for (int i = d; i < CH * 28; i += DI) {
            const int st = i / 28, col = i % 28;
            async_cp_b32(&XD[(size_t)(b * L_ + l0 + c0 + st) * XST + col], &sb[i]);
        }
        __builtin_amdgcn_s_wait_asynccnt(0);
#else
        for (int i = d; i < CH * 28; i += DI) {
            const int st = i / 28, col = i % 28;
            sb[i] = XD[(size_t)(b * L_ + l0 + c0 + st) * XST + col];
        }
#endif
        if (c0 + CH < SEG) {                    // warm L2/near cache for next chunk
            __builtin_prefetch(&XD[(size_t)(b * L_ + l0 + c0 + CH) * XST], 0, 0);
            __builtin_prefetch(&X[(size_t)(b * L_ + l0 + c0 + CH) * DI + d], 0, 0);
        }
        __syncthreads();

        for (int st = 0; st < CH; ++st) {
            const int l = l0 + c0 + st;
            const float u = X[(size_t)(b * L_ + l) * DI + d];
            const float* row = sb + st * 28;
            float del = dbias;
#pragma unroll
            for (int r = 0; r < R_; ++r) del = fmaf(row[r], w[r], del);
            del = softplus_f(del);
            const float* Bt = row + 12;
#pragma unroll
            for (int n = 0; n < N_; ++n) {
                const float dA = __expf(del * A[n]);
                xs[n] = fmaf(dA, xs[n], del * Bt[n] * u);
                pr[n] *= dA;
            }
        }
    }

    const size_t base = ((size_t)bid * DI + d) * N_;
#pragma unroll
    for (int n = 0; n < N_; ++n) { seg_end[base + n] = xs[n]; seg_prod[base + n] = pr[n]; }
}

// ---------------------------------------------------------------------------
// K3: chain segment carries:  x_in[p+1] = end[p] + prod[p] * x_in[p]
// one thread per (s,b,d,n) = 49152 threads
// ---------------------------------------------------------------------------
__global__ void __launch_bounds__(256)
xmamba_carry(const float* __restrict__ seg_end, const float* __restrict__ seg_prod,
             float* __restrict__ carry) {
    const int t = blockIdx.x * blockDim.x + threadIdx.x;
    if (t >= 2 * B_ * DI * N_) return;
    const int n  = t % N_;
    const int d  = (t / N_) % DI;
    const int sb = t / (N_ * DI);               // s*B + b
    float c = 0.f;
#pragma unroll
    for (int p = 0; p < P_; ++p) {
        const size_t idx = ((size_t)(sb * P_ + p) * DI + d) * N_ + n;
        carry[idx] = c;
        c = seg_end[idx] + seg_prod[idx] * c;
    }
}

// ---------------------------------------------------------------------------
// K4: pass 2 — scan with carries, cross-stream C, y = C.x + u*D, fused LN.
// block = (stream, batch, segment); 384 threads = full channel row -> LN local.
// ---------------------------------------------------------------------------
__global__ void __launch_bounds__(DI)
xmamba_scan2(const float* __restrict__ x1, const float* __restrict__ x2,
             const float* __restrict__ dtW1, const float* __restrict__ dtW2,
             const float* __restrict__ dtb1, const float* __restrict__ dtb2,
             const float* __restrict__ Al1,  const float* __restrict__ Al2,
             const float* __restrict__ D1v,  const float* __restrict__ D2v,
             const float* __restrict__ g1,   const float* __restrict__ b1v,
             const float* __restrict__ g2,   const float* __restrict__ b2v,
             const float* __restrict__ xd1,  const float* __restrict__ xd2,
             const float* __restrict__ carry, float* __restrict__ out) {
    const int bid = blockIdx.x;
    const int p   = bid % P_;
    const int b   = (bid / P_) % B_;
    const int s   = bid / (P_ * B_);
    const int d   = threadIdx.x;

    const float* __restrict__ X   = s ? x2  : x1;
    const float* __restrict__ XDo = s ? xd2 : xd1;   // own stream: dt, B
    const float* __restrict__ XDx = s ? xd1 : xd2;   // other stream: C (cross)
    const float* __restrict__ wdt = (s ? dtW2 : dtW1) + (size_t)d * R_;
    const float  dbias            = (s ? dtb2 : dtb1)[d];
    const float* __restrict__ alg = (s ? Al2 : Al1) + (size_t)d * N_;
    const float  Dd = (s ? D2v : D1v)[d];
    const float  gd = (s ? g2  : g1 )[d];
    const float  bd = (s ? b2v : b1v)[d];

    float A[N_], xs[N_], w[R_];
    const size_t cbase = ((size_t)bid * DI + d) * N_;
#pragma unroll
    for (int n = 0; n < N_; ++n) { A[n] = -__expf(alg[n]); xs[n] = carry[cbase + n]; }
#pragma unroll
    for (int r = 0; r < R_; ++r) w[r] = wdt[r];

    __shared__ float sb2[CH * 44];              // dt(12)+B(16) own, C(16) other
    __shared__ float red1[DI / 32], red2[DI / 32], stats[2];

    float* __restrict__ out_s = out + (size_t)s * B_ * L_ * DI;
    const int l0 = p * SEG;

    for (int c0 = 0; c0 < SEG; c0 += CH) {
        __syncthreads();
#if USE_ASYNC_LDS
        for (int i = d; i < CH * 44; i += DI) {
            const int st = i / 44, col = i % 44;
            const float* __restrict__ src = (col < 28) ? XDo : XDx;
            async_cp_b32(&src[(size_t)(b * L_ + l0 + c0 + st) * XST + col], &sb2[i]);
        }
        __builtin_amdgcn_s_wait_asynccnt(0);
#else
        for (int i = d; i < CH * 44; i += DI) {
            const int st = i / 44, col = i % 44;
            const float* __restrict__ src = (col < 28) ? XDo : XDx;
            sb2[i] = src[(size_t)(b * L_ + l0 + c0 + st) * XST + col];
        }
#endif
        if (c0 + CH < SEG) {
            __builtin_prefetch(&XDo[(size_t)(b * L_ + l0 + c0 + CH) * XST], 0, 0);
            __builtin_prefetch(&XDx[(size_t)(b * L_ + l0 + c0 + CH) * XST], 0, 0);
            __builtin_prefetch(&X[(size_t)(b * L_ + l0 + c0 + CH) * DI + d], 0, 0);
        }
        __syncthreads();

        for (int st = 0; st < CH; ++st) {
            const int l = l0 + c0 + st;
            const float u = X[(size_t)(b * L_ + l) * DI + d];
            const float* row = sb2 + st * 44;
            float del = dbias;
#pragma unroll
            for (int r = 0; r < R_; ++r) del = fmaf(row[r], w[r], del);
            del = softplus_f(del);
            const float* Bt = row + 12;
            const float* Ct = row + 28;
            float y = u * Dd;
#pragma unroll
            for (int n = 0; n < N_; ++n) {
                const float dA = __expf(del * A[n]);
                xs[n] = fmaf(dA, xs[n], del * Bt[n] * u);
                y = fmaf(xs[n], Ct[n], y);
            }

            // ---- fused LayerNorm across the 384 channels of this timestep ----
            float s1 = y, s2 = y * y;
#pragma unroll
            for (int o = 16; o > 0; o >>= 1) {
                s1 += __shfl_xor(s1, o, 32);
                s2 += __shfl_xor(s2, o, 32);
            }
            const int wv = d >> 5, ln = d & 31;
            if (ln == 0) { red1[wv] = s1; red2[wv] = s2; }
            __syncthreads();
            if (d == 0) {
                float t1 = 0.f, t2 = 0.f;
#pragma unroll
                for (int q = 0; q < DI / 32; ++q) { t1 += red1[q]; t2 += red2[q]; }
                const float mu  = t1 * (1.f / DI);
                const float var = t2 * (1.f / DI) - mu * mu;
                stats[0] = mu;
                stats[1] = rsqrtf(var + LNEPS);
            }
            __syncthreads();
            // write-once output: keep it out of near caches
            __builtin_nontemporal_store((y - stats[0]) * stats[1] * gd + bd,
                                        &out_s[(size_t)(b * L_ + l) * DI + d]);
        }
    }
}

// ---------------------------------------------------------------------------
extern "C" void kernel_launch(void* const* d_in, const int* in_sizes, int n_in,
                              void* d_out, int out_size, void* d_ws, size_t ws_size,
                              hipStream_t stream) {
    const float* x1   = (const float*)d_in[0];   // x_rgb
    const float* x2   = (const float*)d_in[1];   // x_e
    const float* W1   = (const float*)d_in[2];   // Wxp1 (44,384)
    const float* W2   = (const float*)d_in[3];   // Wxp2
    const float* dtW1 = (const float*)d_in[4];   // (384,12)
    const float* dtW2 = (const float*)d_in[5];
    const float* dtb1 = (const float*)d_in[6];
    const float* dtb2 = (const float*)d_in[7];
    const float* Al1  = (const float*)d_in[8];   // A_log (384,16)
    const float* Al2  = (const float*)d_in[9];
    const float* D1v  = (const float*)d_in[10];
    const float* D2v  = (const float*)d_in[11];
    const float* g1   = (const float*)d_in[12];
    const float* b1v  = (const float*)d_in[13];
    const float* g2   = (const float*)d_in[14];
    const float* b2v  = (const float*)d_in[15];
    float* out = (float*)d_out;

    // workspace layout (floats): xd1 | xd2 | seg_end | seg_prod | carry
    // total = 2*393216 + 3*786432 = 3,145,728 floats = 12.6 MB
    float* ws       = (float*)d_ws;
    float* xd1      = ws;
    float* xd2      = ws + XD_ELEMS;
    float* seg_end  = ws + 2 * XD_ELEMS;
    float* seg_prod = seg_end + CAR_ELEMS;
    float* carry    = seg_prod + CAR_ELEMS;

    // K1: 3072 tile-waves, 4 waves per 128-thread block
    xmamba_proj<<<768, 128, 0, stream>>>(x1, x2, W1, W2, xd1, xd2);

    // K2: 2 streams * 4 batches * 16 segments = 128 blocks of 384 threads
    xmamba_scan1<<<2 * B_ * P_, DI, 0, stream>>>(x1, x2, dtW1, dtW2, dtb1, dtb2,
                                                 Al1, Al2, xd1, xd2,
                                                 seg_end, seg_prod);

    // K3: 49152 threads
    xmamba_carry<<<(2 * B_ * DI * N_ + 255) / 256, 256, 0, stream>>>(seg_end, seg_prod, carry);

    // K4: 128 blocks of 384 threads, fused LayerNorm, writes d_out
    xmamba_scan2<<<2 * B_ * P_, DI, 0, stream>>>(x1, x2, dtW1, dtW2, dtb1, dtb2,
                                                 Al1, Al2, D1v, D2v, g1, b1v, g2, b2v,
                                                 xd1, xd2, carry, out);
}
